// DotAttention_45792941310314
// MI455X (gfx1250) — compile-verified, matrix operand order checked by reading, not run on hardware
//
#include <hip/hip_runtime.h>

typedef __attribute__((ext_vector_type(16))) __bf16 v16bf;
typedef __attribute__((ext_vector_type(8)))  float  v8f;
typedef __attribute__((ext_vector_type(8)))  __bf16 bf16x8;

#define NB    32
#define TDEC  2048
#define TENC  2048
#define DQK   512
#define DV    512

#define SCALE 0.044194173824159216f   // 1/sqrt(512)
#define WCOEF 0.03125f                // 1/(2*sigma^2), sigma=4; TDEC/TENC == 1

#define SST 2052                      // padded f32 row stride (bank spread: 4/row)
#define QST 528                       // padded bf16 row stride (16B aligned rows)

__device__ __forceinline__ void cvt4(v16bf& v, const float4 f, const int base) {
  v[base + 0] = (__bf16)f.x; v[base + 1] = (__bf16)f.y;
  v[base + 2] = (__bf16)f.z; v[base + 3] = (__bf16)f.w;
}

__global__ __launch_bounds__(256) void DotAttention_fused_wmma(
    const float* __restrict__ Q, const float* __restrict__ K,
    const float* __restrict__ V, float* __restrict__ outSum,
    float* __restrict__ outScr)
{
  __shared__ float  slds[16 * SST];     // logits -> probs, f32
  __shared__ __bf16 qlds[16 * QST];     // Q tile, bf16
  __shared__ float  red[2][256];        // softmax partials

  const int tid  = threadIdx.x;
  const int lane = tid & 31;
  const int wave = tid >> 5;
  const int b    = blockIdx.x >> 7;             // 128 i-tiles per batch
  const int i0   = (blockIdx.x & 127) << 4;

  const int nn = lane & 15;                     // A row / B-D col within tile
  const int hk = (lane >> 4) << 3;              // k-group offset: 0 or 8

  // ---- stage Q tile (16 x 512) as bf16 into LDS ----
  {
    const float* qp = Q + ((size_t)b * TDEC + i0) * DQK;
    for (int idx = tid; idx < 16 * DQK; idx += 256) {
      const int r = idx >> 9, c = idx & (DQK - 1);
      qlds[r * QST + c] = (__bf16)qp[r * DQK + c];
    }
  }
  __syncthreads();

  // ---- sweep 1: logits = (Q K^T) * scale - window  ->  LDS ----
  // wave w owns key tiles [w*16, w*16+16); process 2 tiles at once to
  // break the WMMA accumulator dependency chain.
  for (int n = 0; n < 16; n += 2) {
    const int j0a = ((wave << 4) + n) << 4;
    const int j0b = j0a + 16;
    v8f acc0 = {}, acc1 = {};
    const float*  kb0 = K + ((size_t)b * TENC + j0a + nn) * DQK + hk;
    const float*  kb1 = K + ((size_t)b * TENC + j0b + nn) * DQK + hk;
    const __bf16* qr  = &qlds[nn * QST + hk];
    for (int dc = 0; dc < 16; ++dc) {
      const int d0 = dc << 5;
      v16bf a;
      {
        bf16x8 lo = *(const bf16x8*)(qr + d0);
        bf16x8 hi = *(const bf16x8*)(qr + d0 + 16);
#pragma unroll
        for (int t = 0; t < 8; ++t) { a[t] = lo[t]; a[t + 8] = hi[t]; }
      }
      v16bf b0, b1;
      {
        const float* kp = kb0 + d0;
        cvt4(b0, *(const float4*)(kp),      0);
        cvt4(b0, *(const float4*)(kp + 4),  4);
        cvt4(b0, *(const float4*)(kp + 16), 8);
        cvt4(b0, *(const float4*)(kp + 20), 12);
      }
      {
        const float* kp = kb1 + d0;
        cvt4(b1, *(const float4*)(kp),      0);
        cvt4(b1, *(const float4*)(kp + 4),  4);
        cvt4(b1, *(const float4*)(kp + 16), 8);
        cvt4(b1, *(const float4*)(kp + 20), 12);
      }
      acc0 = __builtin_amdgcn_wmma_f32_16x16x32_bf16(false, a, false, b0,
                                                     (short)0, acc0, false, false);
      acc1 = __builtin_amdgcn_wmma_f32_16x16x32_bf16(false, a, false, b1,
                                                     (short)0, acc1, false, false);
    }
    const float jfa = (float)(j0a + nn);
    const float jfb = (float)(j0b + nn);
#pragma unroll
    for (int v = 0; v < 8; ++v) {
      const int m = v + hk;
      const float ibias = (float)(i0 + m);
      slds[m * SST + j0a + nn] = acc0[v] * SCALE - WCOEF * (ibias - jfa);
      slds[m * SST + j0b + nn] = acc1[v] * SCALE - WCOEF * (ibias - jfb);
    }
  }
  __syncthreads();

  // ---- softmax over each of the 16 rows (2048 cols); write scores out ----
  {
    const int r = tid >> 4, c = tid & 15;
    float* row = &slds[r * SST];
    float m = -3.0e38f;
    for (int k = c; k < TENC; k += 16) m = fmaxf(m, row[k]);
    red[0][tid] = m;
    __syncthreads();
    float rm = red[0][r << 4];
#pragma unroll
    for (int t = 1; t < 16; ++t) rm = fmaxf(rm, red[0][(r << 4) + t]);
    float s = 0.f;
    for (int k = c; k < TENC; k += 16) {
      const float e = __expf(row[k] - rm);
      row[k] = e; s += e;
    }
    red[1][tid] = s;
    __syncthreads();
    float rs = 0.f;
#pragma unroll
    for (int t = 0; t < 16; ++t) rs += red[1][(r << 4) + t];
    const float inv = 1.0f / rs;
    float* gs = outScr + ((size_t)b * TDEC + i0 + r) * TENC;
    for (int k = c; k < TENC; k += 16) {
      const float p = row[k] * inv;
      row[k] = p;
      gs[k] = p;
    }
  }
  __syncthreads();

  // ---- sweep 2: summaries = P @ V ; wave w owns cols [w*64, w*64+64) ----
  {
    v8f acc[4] = {};
    const int vcol = (wave << 6) + nn;
    for (int jc = 0; jc < 64; ++jc) {
      const int j0 = jc << 5;
      v16bf a;
      {
        const float* pr = &slds[nn * SST + j0 + hk];
        cvt4(a, *(const float4*)(pr),      0);
        cvt4(a, *(const float4*)(pr + 4),  4);
        cvt4(a, *(const float4*)(pr + 16), 8);
        cvt4(a, *(const float4*)(pr + 20), 12);
      }
      const float* vp = V + ((size_t)b * TENC + j0 + hk) * DV + vcol;
#pragma unroll
      for (int q = 0; q < 4; ++q) {
        v16bf bb;
        const float* vq = vp + (q << 4);
#pragma unroll
        for (int t = 0; t < 8; ++t) {
          bb[t]     = (__bf16)vq[t * DV];
          bb[t + 8] = (__bf16)vq[(t + 16) * DV];
        }
        acc[q] = __builtin_amdgcn_wmma_f32_16x16x32_bf16(false, a, false, bb,
                                                         (short)0, acc[q], false, false);
      }
    }
    float* os = outSum + ((size_t)b * TDEC + i0) * DV;
#pragma unroll
    for (int q = 0; q < 4; ++q) {
#pragma unroll
      for (int v = 0; v < 8; ++v) {
        os[(size_t)(v + hk) * DV + (wave << 6) + (q << 4) + nn] = acc[q][v];
      }
    }
  }
}

extern "C" void kernel_launch(void* const* d_in, const int* in_sizes, int n_in,
                              void* d_out, int out_size, void* d_ws, size_t ws_size,
                              hipStream_t stream) {
  (void)in_sizes; (void)n_in; (void)out_size; (void)d_ws; (void)ws_size;
  const float* Q = (const float*)d_in[0];
  const float* K = (const float*)d_in[1];
  const float* V = (const float*)d_in[2];
  float* outSum = (float*)d_out;
  float* outScr = outSum + (size_t)NB * TDEC * DV;
  dim3 grid(NB * (TDEC / 16));   // 4096 blocks: one (batch, 16-row) tile each
  DotAttention_fused_wmma<<<grid, 256, 0, stream>>>(Q, K, V, outSum, outScr);
}
